// MultiHeadSelfAttention_32624571580467
// MI455X (gfx1250) — matrix operand order repaired and checked by reading, not compiled, not run on hardware
//
#include <hip/hip_runtime.h>

// ---------------------------------------------------------------------------
// MultiHeadSelfAttention for MI455X (gfx1250): wave32, WMMA bf16 (f32 accum),
// async global->LDS tile streaming (ASYNCcnt), double-buffered flash attention.
//   B=2, S=2048, E=1024, H=16, D=64, WINDOW=128 (half = 64)
// Pipeline:
//   1) qkv projections (WMMA, K pre-transposed to [B,H,D,S])
//   2) Wo transpose -> bf16 (one-off)
//   3) flash attention (WMMA, online softmax, async double-buffered K/V)
//   4) output projection (WMMA, async double-buffered tiles) + bias
// ---------------------------------------------------------------------------

#define EMBED   1024
#define HEADS   16
#define HDIM    64
#define BATCH   2
#define SEQ     2048
#define WINHALF 64
#define SCALE   0.03125f   // 1/sqrt(EMBED)

typedef __attribute__((ext_vector_type(16))) __bf16 v16bf;
typedef __attribute__((ext_vector_type(8)))  __bf16 v8bf;
typedef __attribute__((ext_vector_type(8)))  float  v8f;
typedef __attribute__((ext_vector_type(4)))  float  v4f;

union BF16x16 { v16bf v; v8bf h[2]; };

__device__ __forceinline__ v8f wmma_bf16(v16bf a, v16bf b, v8f c) {
  // emits v_wmma_f32_16x16x32_bf16
  return __builtin_amdgcn_wmma_f32_16x16x32_bf16(
      false, a, false, b, (short)0, c, false, false);
}

// --- CDNA5 async global->LDS copy (ASYNCcnt-tracked), 16B per op -----------
__device__ __forceinline__ void async_b128(void* lds_dst, const void* gsrc) {
  // Low 32 bits of a generic LDS-aperture address are the LDS byte offset.
  unsigned l = (unsigned)(unsigned long long)lds_dst;
  unsigned long long g = (unsigned long long)gsrc;
  asm volatile("global_load_async_to_lds_b128 %0, %1, off"
               :: "v"(l), "v"(g) : "memory");
}
__device__ __forceinline__ void wait_async0() {
  asm volatile("s_wait_asynccnt 0x0" ::: "memory");
}

// A-matrix fragment (16x32 bf16) from a row-major [16 x 64] LDS tile.
// lane<16: row M=lane, K in {0..7,16..23}; lane>=16: row M=lane-16,
// K in {8..15,24..31}; kchunk selects K 0-31 / 32-63.   (ISA 7.12.2)
__device__ __forceinline__ v16bf load_a_frag(const __bf16* tile, int ld,
                                             int lane, int kchunk) {
  int m     = lane & 15;
  int kbase = kchunk * 32 + ((lane & 16) ? 8 : 0);
  BF16x16 r;
  r.h[0] = *(const v8bf*)(tile + m * ld + kbase);
  r.h[1] = *(const v8bf*)(tile + m * ld + kbase + 16);
  return r.v;
}

// B-matrix fragment (32x16 bf16) from a row-major [K x N] LDS tile.
// Lane l holds column N = n0 + l%16, K in {0..7,16..23} (l<16) or {8..15,24..31}.
__device__ __forceinline__ v16bf load_b_frag(const __bf16* tile, int ld,
                                             int lane, int kchunk, int n0) {
  int kbase = kchunk * 32 + ((lane & 16) ? 8 : 0);
  int col = n0 + (lane & 15);
  BF16x16 r;
#pragma unroll
  for (int s = 0; s < 16; ++s)
    r.v[s] = tile[(kbase + (s < 8 ? s : s + 8)) * ld + col];
  return r.v;
}

// ---------------------------------------------------------------------------
// Kernel A: per-head projection  out[n,l,h,e] = sum_d x[n,l,h,d] * W[e,d]
//   TRANS=0 -> [B,H,S,D] (q, v);  TRANS=1 -> [B,H,D,S] (k, pre-transposed)
// grid (S/64, H, B), 128 threads = 4 waves.
// ---------------------------------------------------------------------------
template <int TRANS>
__global__ __launch_bounds__(128) void proj_kernel(
    const float* __restrict__ x, const float* __restrict__ W,
    __bf16* __restrict__ out) {
  __shared__ __align__(16) char smem[2 * 64 * 64 * 2];
  __bf16* Xt = (__bf16*)smem;              // [l_local][d], 8KB
  __bf16* Wt = Xt + 64 * 64;               // Wt[d][e] = W[e][d], 8KB

  const int lt = blockIdx.x, h = blockIdx.y, n = blockIdx.z;
  const int tid = threadIdx.x, lane = tid & 31, wave = tid >> 5;
  const int l0 = lt * 64;
  const int half16 = (lane >> 4) & 1, nloc = lane & 15;
  const size_t bh = (size_t)(n * HEADS + h);

  for (int i = tid; i < 64 * 64; i += 128) {
    int e = i >> 6, d = i & 63;
    Wt[d * 64 + e] = (__bf16)W[e * 64 + d];
  }
  for (int i = tid; i < 64 * 64; i += 128) {
    int r = i >> 6, d = i & 63;
    Xt[i] = (__bf16)x[((size_t)(n * SEQ + l0 + r)) * EMBED + h * HDIM + d];
  }
  __syncthreads();

  const __bf16* atile = Xt + wave * 16 * 64;
  v16bf a0 = load_a_frag(atile, 64, lane, 0);
  v16bf a1 = load_a_frag(atile, 64, lane, 1);

  v8f acc[4];
#pragma unroll
  for (int t = 0; t < 4; ++t) {
    v8f c = {};
    c = wmma_bf16(a0, load_b_frag(Wt, 64, lane, 0, t * 16), c);
    c = wmma_bf16(a1, load_b_frag(Wt, 64, lane, 1, t * 16), c);
    acc[t] = c;
  }

  if (TRANS) {
    // C-fragment rows are contiguous in l per lane -> one b128 store per tile.
#pragma unroll
    for (int t = 0; t < 4; ++t) {
      v8bf pk;
#pragma unroll
      for (int c = 0; c < 8; ++c) pk[c] = (__bf16)acc[t][c];
      int e = t * 16 + nloc;
      *(v8bf*)(out + (bh * HDIM + e) * SEQ + l0 + wave * 16 + 8 * half16) = pk;
    }
  } else {
    // Bounce through LDS for fully coalesced b128 row stores.
    __syncthreads();  // everyone done reading Xt/Wt
#pragma unroll
    for (int t = 0; t < 4; ++t)
#pragma unroll
      for (int c = 0; c < 8; ++c)
        Xt[(wave * 16 + c + 8 * half16) * 64 + t * 16 + nloc] =
            (__bf16)acc[t][c];
    __syncthreads();
    for (int i = tid * 8; i < 64 * 64; i += 128 * 8) {
      int r = i >> 6, d = i & 63;
      *(v8bf*)(out + (bh * SEQ + l0 + r) * HDIM + d) = *(const v8bf*)(Xt + i);
    }
  }
}

// ---------------------------------------------------------------------------
// Kernel B: WoT[k][n] = bf16(Wo[n][k])   (one-off, feeds oproj B-fragments)
// ---------------------------------------------------------------------------
__global__ __launch_bounds__(256) void wo_transpose_kernel(
    const float* __restrict__ Wo, __bf16* __restrict__ WoT) {
  int idx = blockIdx.x * 256 + threadIdx.x;   // k*1024 + n
  int k = idx >> 10, n = idx & 1023;
  WoT[idx] = (__bf16)Wo[n * EMBED + k];
}

// ---------------------------------------------------------------------------
// Kernel C: flash attention, faithful to the reference's mask semantics:
//   energy = -1e20 where (|q-k| <= 64) && (mask[b,0,q,k] == 0); else raw.
//   softmax(energy/32); O = P @ V -> bf16 [B,S,E].
// Async double-buffered K/V tiles; grid (S/64, H, B); 128 threads = 4 waves.
// ---------------------------------------------------------------------------
__global__ __launch_bounds__(128) void attn_kernel(
    const __bf16* __restrict__ qb,   // [B,H,S,D]
    const __bf16* __restrict__ kb,   // [B,H,D,S]
    const __bf16* __restrict__ vb,   // [B,H,S,D]
    const float*  __restrict__ mask, // [B,1,S,S]
    __bf16* __restrict__ attn_out) { // [B,S,E] bf16
  __shared__ __align__(16) char smem[48 * 1024];
  __bf16* Qs    = (__bf16*)smem;               // 8KB [q][d]
  __bf16* Kbase = (__bf16*)(smem + 8192);      // 2 x 4096 elems, [d][k]
  __bf16* Vbase = (__bf16*)(smem + 24576);     // 2 x 4096 elems, [k][e]
  __bf16* Ps    = (__bf16*)(smem + 40960);     // 4 waves x [16][64]
  __bf16* Ob    = (__bf16*)(smem + 8192);      // epilogue reuse

  const int qt = blockIdx.x, h = blockIdx.y, n = blockIdx.z;
  const int tid = threadIdx.x, lane = tid & 31, wave = tid >> 5;
  const int q0 = qt * 64;
  const int half16 = (lane >> 4) & 1, nloc = lane & 15;
  const size_t bh = (size_t)(n * HEADS + h);
  const __bf16* qptr = qb + bh * SEQ * HDIM;
  const __bf16* kptr = kb + bh * HDIM * SEQ;
  const __bf16* vptr = vb + bh * SEQ * HDIM;

  // Q tile + KV tile 0: async into LDS
  for (int i = tid * 8; i < 64 * 64; i += 128 * 8)
    async_b128(Qs + i, qptr + (size_t)q0 * HDIM + i);
  for (int i = tid * 8; i < 64 * 64; i += 128 * 8) {
    int d = i >> 6, k = i & 63;
    async_b128(Kbase + i, kptr + (size_t)d * SEQ + k);
    async_b128(Vbase + i, vptr + i);
  }
  wait_async0();
  __syncthreads();

  v16bf aq0 = load_a_frag(Qs + wave * 16 * 64, 64, lane, 0);
  v16bf aq1 = load_a_frag(Qs + wave * 16 * 64, 64, lane, 1);

  v8f o[4] = {};
  float mrow[8], lrow[8];
#pragma unroll
  for (int c = 0; c < 8; ++c) { mrow[c] = -3.0e38f; lrow[c] = 0.0f; }

  for (int kt = 0; kt < SEQ / 64; ++kt) {
    const int cur = (kt & 1) * 4096, nxt = ((kt + 1) & 1) * 4096;
    if (kt + 1 < SEQ / 64) {  // stream next tile while computing this one
      for (int i = tid * 8; i < 64 * 64; i += 128 * 8) {
        int d = i >> 6, k = i & 63;
        async_b128(Kbase + nxt + i,
                   kptr + (size_t)d * SEQ + (kt + 1) * 64 + k);
        async_b128(Vbase + nxt + i,
                   vptr + (size_t)((kt + 1) * 64) * HDIM + i);
      }
    }
    const __bf16* Ks = Kbase + cur;
    const __bf16* Vs = Vbase + cur;

    // ---- energy = Q (16x64) x K^T tile (64x64) ----
    v8f e[4];
#pragma unroll
    for (int t = 0; t < 4; ++t) {
      v8f acc = {};
      acc = wmma_bf16(aq0, load_b_frag(Ks, 64, lane, 0, t * 16), acc);
      acc = wmma_bf16(aq1, load_b_frag(Ks, 64, lane, 1, t * 16), acc);
      e[t] = acc;
    }

    // ---- mask fill only where the tile can intersect the window ----
    if (kt * 64 <= q0 + 63 + WINHALF && kt * 64 + 63 >= q0 - WINHALF) {
#pragma unroll
      for (int t = 0; t < 4; ++t)
#pragma unroll
        for (int c = 0; c < 8; ++c) {
          int qg = q0 + wave * 16 + c + 8 * half16;
          int kg = kt * 64 + t * 16 + nloc;
          int dd = qg - kg;
          if (dd <= WINHALF && dd >= -WINHALF) {
            if (mask[((size_t)n * SEQ + qg) * SEQ + kg] == 0.0f)
              e[t][c] = -1e20f;  // combined == 0 -> fill
          }
        }
    }
#pragma unroll
    for (int t = 0; t < 4; ++t)
#pragma unroll
      for (int c = 0; c < 8; ++c) e[t][c] *= SCALE;

    // ---- online softmax (row reductions across the 16-lane half) ----
#pragma unroll
    for (int c = 0; c < 8; ++c) {
      float tm = fmaxf(fmaxf(e[0][c], e[1][c]), fmaxf(e[2][c], e[3][c]));
      tm = fmaxf(tm, __shfl_xor(tm, 8, 32));
      tm = fmaxf(tm, __shfl_xor(tm, 4, 32));
      tm = fmaxf(tm, __shfl_xor(tm, 2, 32));
      tm = fmaxf(tm, __shfl_xor(tm, 1, 32));
      float mnew  = fmaxf(mrow[c], tm);
      float alpha = __expf(mrow[c] - mnew);
      mrow[c] = mnew;
      float rs = 0.0f;
#pragma unroll
      for (int t = 0; t < 4; ++t) {
        float p = __expf(e[t][c] - mnew);
        e[t][c] = p;
        rs += p;
      }
      rs += __shfl_xor(rs, 8, 32);
      rs += __shfl_xor(rs, 4, 32);
      rs += __shfl_xor(rs, 2, 32);
      rs += __shfl_xor(rs, 1, 32);
      lrow[c] = lrow[c] * alpha + rs;
#pragma unroll
      for (int t = 0; t < 4; ++t) o[t][c] *= alpha;
    }

    // ---- P (C layout) -> per-wave LDS -> A layout; O += P @ V ----
    __bf16* pw = Ps + wave * 16 * 64;
#pragma unroll
    for (int t = 0; t < 4; ++t)
#pragma unroll
      for (int c = 0; c < 8; ++c)
        pw[(c + 8 * half16) * 64 + t * 16 + nloc] = (__bf16)e[t][c];
    // same-wave LDS RAW: DS ops are issue-ordered per wave; no barrier needed.
    v16bf ap0 = load_a_frag(pw, 64, lane, 0);
    v16bf ap1 = load_a_frag(pw, 64, lane, 1);
#pragma unroll
    for (int t = 0; t < 4; ++t) {
      o[t] = wmma_bf16(ap0, load_b_frag(Vs, 64, lane, 0, t * 16), o[t]);
      o[t] = wmma_bf16(ap1, load_b_frag(Vs, 64, lane, 1, t * 16), o[t]);
    }

    wait_async0();     // this wave's next-tile asyncs have landed in LDS
    __syncthreads();   // everyone done with cur + everyone's loads visible
  }

  // ---- normalize, bounce through LDS, coalesced bf16 stores ----
#pragma unroll
  for (int t = 0; t < 4; ++t) {
#pragma unroll
    for (int c = 0; c < 8; ++c) {
      float inv = __builtin_amdgcn_rcpf(lrow[c]);
      Ob[(wave * 16 + c + 8 * half16) * 64 + t * 16 + nloc] =
          (__bf16)(o[t][c] * inv);
    }
  }
  __syncthreads();
  for (int i = tid * 8; i < 64 * 64; i += 128 * 8) {
    int r = i >> 6, d = i & 63;
    *(v8bf*)(attn_out + ((size_t)(n * SEQ + q0 + r)) * EMBED + h * HDIM + d) =
        *(const v8bf*)(Ob + i);
  }
}

// ---------------------------------------------------------------------------
// Kernel D: out[m,n] = sum_k A[m,k]*WoT[k,n] + bo[n]; M=B*S, K=N=E.
// Async double-buffered A/B tiles. grid (M/64, N/64), 128 threads = 4 waves.
// ---------------------------------------------------------------------------
__global__ __launch_bounds__(128) void oproj_kernel(
    const __bf16* __restrict__ A,     // [M, E] bf16 (attention output)
    const __bf16* __restrict__ WoT,   // [E, E] bf16, WoT[k][n] = Wo[n][k]
    const float*  __restrict__ bo,    // [E]
    float* __restrict__ out) {        // [M, E] f32
  __shared__ __align__(16) char smem[32 * 1024];
  __bf16* Abase = (__bf16*)smem;               // 2 x 4096 elems
  __bf16* Bbase = (__bf16*)(smem + 16384);     // 2 x 4096 elems
  float*  Cs    = (float*)smem;                // 16KB epilogue reuse

  const int m0 = blockIdx.x * 64, n0 = blockIdx.y * 64;
  const int tid = threadIdx.x, lane = tid & 31, wave = tid >> 5;
  const int half16 = (lane >> 4) & 1, nloc = lane & 15;
  v8f acc[4] = {};

  for (int i = tid * 8; i < 64 * 64; i += 128 * 8) {
    int r = i >> 6, d = i & 63;
    async_b128(Abase + i, A + (size_t)(m0 + r) * EMBED + d);
    async_b128(Bbase + i, WoT + (size_t)r * EMBED + n0 + d);
  }
  wait_async0();
  __syncthreads();

  for (int kc = 0; kc < EMBED / 64; ++kc) {
    const int cur = (kc & 1) * 4096, nxt = ((kc + 1) & 1) * 4096;
    if (kc + 1 < EMBED / 64) {
      for (int i = tid * 8; i < 64 * 64; i += 128 * 8) {
        int r = i >> 6, d = i & 63;
        async_b128(Abase + nxt + i,
                   A + (size_t)(m0 + r) * EMBED + (kc + 1) * 64 + d);
        async_b128(Bbase + nxt + i,
                   WoT + (size_t)((kc + 1) * 64 + r) * EMBED + n0 + d);
      }
    }
    const __bf16* As = Abase + cur;
    const __bf16* Bs = Bbase + cur;
    v16bf a0 = load_a_frag(As + wave * 16 * 64, 64, lane, 0);
    v16bf a1 = load_a_frag(As + wave * 16 * 64, 64, lane, 1);
#pragma unroll
    for (int t = 0; t < 4; ++t) {
      acc[t] = wmma_bf16(a0, load_b_frag(Bs, 64, lane, 0, t * 16), acc[t]);
      acc[t] = wmma_bf16(a1, load_b_frag(Bs, 64, lane, 1, t * 16), acc[t]);
    }
    wait_async0();
    __syncthreads();
  }

  // bias + LDS bounce -> coalesced b128 f32 stores
#pragma unroll
  for (int t = 0; t < 4; ++t) {
    float bias = bo[n0 + t * 16 + nloc];
#pragma unroll
    for (int c = 0; c < 8; ++c)
      Cs[(wave * 16 + c + 8 * half16) * 64 + t * 16 + nloc] = acc[t][c] + bias;
  }
  __syncthreads();
  for (int i = tid * 4; i < 64 * 64; i += 128 * 4) {
    int r = i >> 6, d = i & 63;
    *(v4f*)(out + (size_t)(m0 + r) * EMBED + n0 + d) = *(const v4f*)(Cs + i);
  }
}

// ---------------------------------------------------------------------------
extern "C" void kernel_launch(void* const* d_in, const int* in_sizes, int n_in,
                              void* d_out, int out_size, void* d_ws,
                              size_t ws_size, hipStream_t stream) {
  const float* values = (const float*)d_in[0];
  const float* keys   = (const float*)d_in[1];
  const float* query  = (const float*)d_in[2];
  const float* mask   = (const float*)d_in[3];
  const float* Wq     = (const float*)d_in[4];
  const float* Wk     = (const float*)d_in[5];
  const float* Wv     = (const float*)d_in[6];
  const float* Wo     = (const float*)d_in[7];
  const float* bo     = (const float*)d_in[8];
  float* out          = (float*)d_out;

  // workspace layout (all bf16 intermediates)
  const size_t nelem = (size_t)BATCH * HEADS * SEQ * HDIM;  // 4M
  char* ws = (char*)d_ws;
  __bf16* qb      = (__bf16*)(ws);                  // 8MB  [B,H,S,D]
  __bf16* kb      = (__bf16*)(ws + nelem * 2);      // 8MB  [B,H,D,S]
  __bf16* vb      = (__bf16*)(ws + nelem * 4);      // 8MB  [B,H,S,D]
  __bf16* attn_bf = (__bf16*)(ws + nelem * 6);      // 8MB  [B,S,E]
  __bf16* woT     = (__bf16*)(ws + nelem * 8);      // 2MB  [E,E]

  dim3 gridP(SEQ / 64, HEADS, BATCH);
  proj_kernel<0><<<gridP, 128, 0, stream>>>(query,  Wq, qb);
  proj_kernel<1><<<gridP, 128, 0, stream>>>(keys,   Wk, kb);  // pre-transposed
  proj_kernel<0><<<gridP, 128, 0, stream>>>(values, Wv, vb);

  wo_transpose_kernel<<<(EMBED * EMBED) / 256, 256, 0, stream>>>(Wo, woT);

  attn_kernel<<<dim3(SEQ / 64, HEADS, BATCH), 128, 0, stream>>>(qb, kb, vb,
                                                                mask, attn_bf);

  oproj_kernel<<<dim3((BATCH * SEQ) / 64, EMBED / 64), 128, 0, stream>>>(
      attn_bf, woT, bo, out);
}
